// MLA_45784351375731
// MI455X (gfx1250) — matrix operand order repaired and checked by reading, not compiled, not run on hardware
//
#include <hip/hip_runtime.h>
#include <hip/hip_bf16.h>
#include <math.h>
#include <stdint.h>

// MLA forward for MI455X (gfx1250, wave32, WMMA + Tensor Data Mover).
// Shapes: B=2 S=2048 HID=2048 H=16 D=128 R=64 CKV=512 CQ=1536

typedef __attribute__((ext_vector_type(16))) _Float16     v16h;
typedef __attribute__((ext_vector_type(8)))  _Float16     v8h;
typedef __attribute__((ext_vector_type(8)))  float        v8f;
typedef __attribute__((ext_vector_type(4)))  unsigned int u32x4;
typedef __attribute__((ext_vector_type(4)))  int          i32x4;
typedef __attribute__((ext_vector_type(8)))  int          i32x8;

#define B_    2
#define S_    2048
#define HID_  2048
#define H_    16
#define D_    128
#define R_    64
#define CKV_  512
#define CQ_   1536

#define HAS_TDM __has_builtin(__builtin_amdgcn_tensor_load_to_lds)

// ---------------------------------------------------------------------------
// WMMA operand fragments (16-bit, v_wmma_f32_16x16x32_f16), per ISA 7.12.2.
// ---------------------------------------------------------------------------
__device__ __forceinline__ v16h ldfrag(const _Float16* p0, const _Float16* p1) {
    v8h a = *(const v8h*)p0;
    v8h b = *(const v8h*)p1;
    v16h r;
#pragma unroll
    for (int e = 0; e < 8; ++e) { r[e] = a[e]; r[8 + e] = b[e]; }
    return r;
}
// A 16x32: lanes 0-15 hold K {0..7,16..23}; lanes 16-31 hold K {8..15,24..31}.
__device__ __forceinline__ v16h frag_a(const _Float16* rowbase, int hl, int kbase) {
    const _Float16* p0 = rowbase + kbase + hl * 8;
    return ldfrag(p0, p0 + 16);
}
// B 32x16: lanes 0-15 hold K {0..7,16..23}; lanes 16-31 hold K {8..15,24..31}.
__device__ __forceinline__ v16h frag_b(const _Float16* rowbase, int hl, int kbase) {
    const _Float16* p0 = rowbase + kbase + hl * 8;
    return ldfrag(p0, p0 + 16);
}

// ---------------------------------------------------------------------------
// Tensor Data Mover: issue a 2-D tile load (f16 elements) global -> LDS.
// Rows of tile_d0 elements; LDS gets 4 dwords of pad after every 32 dwords
// (tile_d0 = 64 halves = 32 dwords), producing the 72-half padded row stride
// the WMMA fragment loads expect.  D# layout per CDNA5 ISA.
// ---------------------------------------------------------------------------
#if HAS_TDM
__device__ __forceinline__ void tdm_load_2d(unsigned lds_off, const void* gptr,
                                            unsigned tensor_d0, unsigned tensor_d1,
                                            unsigned tile_d0, unsigned tile_d1,
                                            unsigned stride0_elems) {
    unsigned long long ga = (unsigned long long)gptr;
    u32x4 g0;
    g0[0] = 1u;                                             // count=1, user D#
    g0[1] = lds_off;                                        // lds_addr
    g0[2] = (unsigned)(ga & 0xFFFFFFFFu);                   // global_addr[31:0]
    g0[3] = (unsigned)((ga >> 32) & 0x01FFFFFFu) | (2u << 30); // addr[56:32] | type=2
    i32x8 g1;
    g1[0] = (int)((1u << 16)      // data_size = 2 bytes
                | (1u << 20)      // pad_enable
                | (4u << 22)      // pad_interval: 32 dwords
                | (3u << 25));    // pad_amount: 4 dwords
    g1[1] = (int)((tensor_d0 & 0xFFFFu) << 16);             // tensor_dim0 lo16
    g1[2] = (int)((tensor_d0 >> 16) | ((tensor_d1 & 0xFFFFu) << 16));
    g1[3] = (int)((tensor_d1 >> 16) | (tile_d0 << 16));     // tile_dim0
    g1[4] = (int)tile_d1;                                   // tile_dim1 (tile_dim2=0)
    g1[5] = (int)stride0_elems;                             // tensor_dim0_stride lo32
    g1[6] = 0;
    g1[7] = 0;
    i32x4 z4 = {0, 0, 0, 0};
#if __clang_major__ >= 23
    i32x8 z8 = {0, 0, 0, 0, 0, 0, 0, 0};
    __builtin_amdgcn_tensor_load_to_lds(g0, g1, z4, z4, z8, 0);
#else
    __builtin_amdgcn_tensor_load_to_lds(g0, g1, z4, z4, 0);
#endif
}
#endif

// ---------------------------------------------------------------------------
// Prep kernels: f32 -> f16 copy, and W[K,N] f32 -> Wt[N,K] f16 transpose.
// ---------------------------------------------------------------------------
__global__ void f32_to_f16(const float* __restrict__ src, _Float16* __restrict__ dst,
                           int n) {
    int i = blockIdx.x * blockDim.x + threadIdx.x;
    if (i < n) dst[i] = (_Float16)src[i];
}

__global__ void __launch_bounds__(256)
transpose_f32_f16(const float* __restrict__ W, _Float16* __restrict__ Wt,
                  int K, int N) {
    __shared__ float tile[32][33];
    const int nb = blockIdx.x * 32, kb = blockIdx.y * 32;
    const int tx = threadIdx.x & 31, ty = threadIdx.x >> 5;
#pragma unroll
    for (int yy = ty; yy < 32; yy += 8)
        tile[yy][tx] = W[(size_t)(kb + yy) * N + (nb + tx)];
    __syncthreads();
#pragma unroll
    for (int yy = ty; yy < 32; yy += 8)
        Wt[(size_t)(nb + yy) * K + (kb + tx)] = (_Float16)tile[tx][yy];
}

// ---------------------------------------------------------------------------
// GEMM: C[M,N] = A[M,K](f16) @ Wt[N,K](f16)^T + bias(f32).
// Block tile 128x64, BK=64, 256 threads (8 waves); wave w owns a 16x64 strip.
// TDM path: double-buffered LDS tiles, wave 0 drives tensor_load_to_lds and
// s_wait_tensorcnt; all waves only touch LDS + WMMA.
// ---------------------------------------------------------------------------
template <typename TC>
__global__ void __launch_bounds__(256)
gemm_wmma(const _Float16* __restrict__ A, const _Float16* __restrict__ Wt,
          const float* __restrict__ bias, TC* __restrict__ C,
          int M, int N, int K) {
    const int tid  = threadIdx.x;
    const int wave = tid >> 5;
    const int lane = tid & 31;
    const int hl   = (lane >> 4) & 1;
    const int l15  = lane & 15;
    const int m0   = blockIdx.y * 128;
    const int n0   = blockIdx.x * 64;

    v8f acc[4] = {};

#if HAS_TDM
    __shared__ __align__(16) _Float16 sA[2][128 * 72];
    __shared__ __align__(16) _Float16 sB[2][64 * 72];

    auto issue = [&](int buf, int k0) {
        tdm_load_2d((unsigned)(size_t)&sA[buf][0], A + (size_t)m0 * K + k0,
                    (unsigned)K, (unsigned)M, 64u, 128u, (unsigned)K);
        tdm_load_2d((unsigned)(size_t)&sB[buf][0], Wt + (size_t)n0 * K + k0,
                    (unsigned)K, (unsigned)N, 64u, 64u, (unsigned)K);
    };

    if (wave == 0) issue(0, 0);
    const int nk = K / 64;
    for (int t = 0; t < nk; ++t) {
        if (wave == 0) __builtin_amdgcn_s_wait_tensorcnt(0);
        __syncthreads();                       // tile t resident for all waves
        if (wave == 0 && t + 1 < nk) issue((t + 1) & 1, (t + 1) * 64);

        const int buf = t & 1;
        const _Float16* arow = &sA[buf][(wave * 16 + l15) * 72];
#pragma unroll
        for (int kt = 0; kt < 2; ++kt) {
            v16h af = frag_a(arow, hl, kt * 32);
#pragma unroll
            for (int j = 0; j < 4; ++j) {
                v16h bf = frag_b(&sB[buf][(j * 16 + l15) * 72], hl, kt * 32);
                acc[j] = __builtin_amdgcn_wmma_f32_16x16x32_f16(
                    false, af, false, bf, (short)0, acc[j], false, false);
            }
        }
        __syncthreads();                       // done reading buf before reuse
    }
#else
    __shared__ __align__(16) _Float16 sA[128 * 72];
    __shared__ __align__(16) _Float16 sB[64 * 72];
    for (int k0 = 0; k0 < K; k0 += 64) {
#pragma unroll 4
        for (int i = tid; i < 128 * 64; i += 256) {
            int r = i >> 6, c = i & 63;
            sA[r * 72 + c] = A[(size_t)(m0 + r) * K + (k0 + c)];
        }
#pragma unroll 4
        for (int i = tid; i < 64 * 64; i += 256) {
            int n = i >> 6, c = i & 63;
            sB[n * 72 + c] = Wt[(size_t)(n0 + n) * K + (k0 + c)];
        }
        __syncthreads();
        const _Float16* arow = &sA[(wave * 16 + l15) * 72];
#pragma unroll
        for (int kt = 0; kt < 2; ++kt) {
            v16h af = frag_a(arow, hl, kt * 32);
#pragma unroll
            for (int j = 0; j < 4; ++j) {
                v16h bf = frag_b(&sB[(j * 16 + l15) * 72], hl, kt * 32);
                acc[j] = __builtin_amdgcn_wmma_f32_16x16x32_f16(
                    false, af, false, bf, (short)0, acc[j], false, false);
            }
        }
        __syncthreads();
    }
#endif

    // C/D layout: lane 0-15 -> N=lane, M=r ; lane 16-31 -> N=lane-16, M=r+8.
#pragma unroll
    for (int j = 0; j < 4; ++j) {
        int col  = n0 + j * 16 + l15;
        float bv = bias[col];
#pragma unroll
        for (int r = 0; r < 8; ++r) {
            int row = m0 + wave * 16 + r + hl * 8;
            C[(size_t)row * N + col] = (TC)(acc[j][r] + bv);
        }
    }
}

// ---------------------------------------------------------------------------
// RoPE + head packing (see round-0 comments).
// ---------------------------------------------------------------------------
__global__ void rope_pack(const _Float16* __restrict__ q_lat, const _Float16* __restrict__ q_rope,
                          const _Float16* __restrict__ k_lat, const _Float16* __restrict__ k_rope,
                          const _Float16* __restrict__ v_lin, const int* __restrict__ pos_ids,
                          _Float16* __restrict__ Qp, _Float16* __restrict__ Kp,
                          _Float16* __restrict__ Vp) {
    const int bs = blockIdx.x;
    const int b  = bs / S_;
    const int s  = bs % S_;
    const size_t row = (size_t)bs;
    const float posq = (float)pos_ids[s];
    const float posk = (float)s;

    for (int i = threadIdx.x; i < H_ * 64; i += blockDim.x) {   // rotated part
        int h = i >> 6, j = i & 63;
        int f = j & 31;
        float invf = __powf(10000.0f, -(float)f * (1.0f / 32.0f));
        size_t src  = row * (size_t)(H_ * D_) + h * D_ + j;
        size_t srcp = row * (size_t)(H_ * D_) + h * D_ + ((j < 32) ? j + 32 : j - 32);
        float sgn = (j < 32) ? -1.0f : 1.0f;
        size_t dst = ((size_t)(b * H_ + h) * S_ + s) * 192 + j;

        float aq = posq * invf;
        Qp[dst] = (_Float16)((float)q_lat[src] * __cosf(aq) +
                             sgn * (float)q_lat[srcp] * __sinf(aq));
        float ak = posk * invf;
        Kp[dst] = (_Float16)((float)k_lat[src] * __cosf(ak) +
                             sgn * (float)k_lat[srcp] * __sinf(ak));
    }
    for (int i = threadIdx.x; i < H_ * 64; i += blockDim.x) {   // pass-through
        int h = i >> 6, j = (i & 63) + 64;
        size_t base = ((size_t)(b * H_ + h) * S_ + s) * 192;
        size_t src  = row * (size_t)(H_ * D_) + h * D_ + j;
        Qp[base + j] = q_lat[src];
        Kp[base + j] = k_lat[src];
    }
    for (int i = threadIdx.x; i < H_ * 64; i += blockDim.x) {   // rope features
        int h = i >> 6, j = i & 63;
        size_t base = ((size_t)(b * H_ + h) * S_ + s) * 192;
        size_t src  = row * (size_t)(H_ * R_) + h * R_ + j;
        Qp[base + 128 + j] = q_rope[src];
        Kp[base + 128 + j] = k_rope[src];
    }
    for (int i = threadIdx.x; i < H_ * D_; i += blockDim.x) {   // V pack
        int h = i >> 7, j = i & 127;
        Vp[((size_t)(b * H_ + h) * S_ + s) * D_ + j] =
            v_lin[row * (size_t)(H_ * D_) + h * D_ + j];
    }
}

// ---------------------------------------------------------------------------
// Flash attention: block = (64 q-rows, head, batch), 4 waves x 16 q-rows.
// 64-key tiles: QK^T via WMMA, online softmax (shfl_xor within 16-lane
// halves), P staged through wave-private LDS (s_wait_dscnt), O += P@V.
// ---------------------------------------------------------------------------
__global__ void __launch_bounds__(128)
flash_attn(const _Float16* __restrict__ Qp, const _Float16* __restrict__ Kp,
           const _Float16* __restrict__ Vp, const float* __restrict__ mask,
           _Float16* __restrict__ Out) {
    __shared__ __align__(16) _Float16 sq[64 * 200];
    __shared__ __align__(16) _Float16 sk[64 * 200];
    __shared__ __align__(16) _Float16 sv[128 * 72];
    __shared__ __align__(16) _Float16 sp[4 * 16 * 72];

    const int tid  = threadIdx.x;
    const int wave = tid >> 5;
    const int lane = tid & 31;
    const int hl   = (lane >> 4) & 1;
    const int l15  = lane & 15;
    const int q0   = blockIdx.x * 64;
    const int h    = blockIdx.y;
    const int b    = blockIdx.z;
    const size_t head = (size_t)(b * H_ + h) * S_;
    const float scale = 0.07216878364870323f;  // 1/sqrt(D+R)

    for (int i = tid; i < 64 * 192; i += 128) {
        int r = i / 192, c = i % 192;
        sq[r * 200 + c] = Qp[(head + q0 + r) * 192 + c];
    }

    v8f o[8] = {};
    float mrow[8], lrow[8];
#pragma unroll
    for (int r = 0; r < 8; ++r) { mrow[r] = -INFINITY; lrow[r] = 0.0f; }

    for (int kb = 0; kb < S_; kb += 64) {
        __syncthreads();
        for (int i = tid; i < 64 * 192; i += 128) {
            int r = i / 192, c = i % 192;
            sk[r * 200 + c] = Kp[(head + kb + r) * 192 + c];
        }
        for (int i = tid; i < 64 * 128; i += 128) {
            int key = i >> 7, d = i & 127;
            sv[d * 72 + key] = Vp[(head + kb + key) * D_ + d];
        }
        __syncthreads();

        v8f acc[4] = {};
        const _Float16* arow = &sq[(wave * 16 + l15) * 200];
#pragma unroll
        for (int kt = 0; kt < 6; ++kt) {
            v16h af = frag_a(arow, hl, kt * 32);
#pragma unroll
            for (int j = 0; j < 4; ++j) {
                v16h bf = frag_b(&sk[(j * 16 + l15) * 200], hl, kt * 32);
                acc[j] = __builtin_amdgcn_wmma_f32_16x16x32_f16(
                    false, af, false, bf, (short)0, acc[j], false, false);
            }
        }

#pragma unroll
        for (int j = 0; j < 4; ++j) {
#pragma unroll
            for (int r = 0; r < 8; ++r) {
                int qrow = q0 + wave * 16 + r + hl * 8;
                int key  = kb + j * 16 + l15;
                acc[j][r] = acc[j][r] * scale + mask[(size_t)qrow * S_ + key];
            }
        }

#pragma unroll
        for (int r = 0; r < 8; ++r) {
            float mx = fmaxf(fmaxf(acc[0][r], acc[1][r]), fmaxf(acc[2][r], acc[3][r]));
            mx = fmaxf(mx, __shfl_xor(mx, 1, 32));
            mx = fmaxf(mx, __shfl_xor(mx, 2, 32));
            mx = fmaxf(mx, __shfl_xor(mx, 4, 32));
            mx = fmaxf(mx, __shfl_xor(mx, 8, 32));
            float mnew = fmaxf(mrow[r], mx);
            float corr = __expf(mrow[r] - mnew);
            float p0 = __expf(acc[0][r] - mnew);
            float p1 = __expf(acc[1][r] - mnew);
            float p2 = __expf(acc[2][r] - mnew);
            float p3 = __expf(acc[3][r] - mnew);
            acc[0][r] = p0; acc[1][r] = p1; acc[2][r] = p2; acc[3][r] = p3;
            float rs = p0 + p1 + p2 + p3;
            rs += __shfl_xor(rs, 1, 32);
            rs += __shfl_xor(rs, 2, 32);
            rs += __shfl_xor(rs, 4, 32);
            rs += __shfl_xor(rs, 8, 32);
            lrow[r] = lrow[r] * corr + rs;
            mrow[r] = mnew;
#pragma unroll
            for (int nt = 0; nt < 8; ++nt) o[nt][r] *= corr;
        }

        _Float16* pw = &sp[wave * 16 * 72];
#pragma unroll
        for (int j = 0; j < 4; ++j)
#pragma unroll
            for (int r = 0; r < 8; ++r)
                pw[(r + hl * 8) * 72 + j * 16 + l15] = (_Float16)acc[j][r];
        asm volatile("s_wait_dscnt 0" ::: "memory");  // wave-local P visibility

#pragma unroll
        for (int kk = 0; kk < 2; ++kk) {
            v16h af = frag_a(&pw[l15 * 72], hl, kk * 32);
#pragma unroll
            for (int nt = 0; nt < 8; ++nt) {
                v16h bf = frag_b(&sv[(nt * 16 + l15) * 72], hl, kk * 32);
                o[nt] = __builtin_amdgcn_wmma_f32_16x16x32_f16(
                    false, af, false, bf, (short)0, o[nt], false, false);
            }
        }
    }

#pragma unroll
    for (int nt = 0; nt < 8; ++nt) {
#pragma unroll
        for (int r = 0; r < 8; ++r) {
            int qrow = q0 + wave * 16 + r + hl * 8;
            float val = o[nt][r] / lrow[r];
            Out[(size_t)(b * S_ + qrow) * (H_ * D_) + h * D_ + nt * 16 + l15] =
                (_Float16)val;
        }
    }
}

// ---------------------------------------------------------------------------
extern "C" void kernel_launch(void* const* d_in, const int* in_sizes, int n_in,
                              void* d_out, int out_size, void* d_ws, size_t ws_size,
                              hipStream_t stream) {
    (void)in_sizes; (void)n_in; (void)out_size; (void)ws_size;

    const float* hidden = (const float*)d_in[0];
    const float* mask   = (const float*)d_in[1];
    const int*   pos    = (const int*)d_in[2];
    const float* W_DKV  = (const float*)d_in[3];  const float* b_DKV = (const float*)d_in[4];
    const float* W_DQ   = (const float*)d_in[5];  const float* b_DQ  = (const float*)d_in[6];
    const float* W_UK   = (const float*)d_in[7];  const float* b_UK  = (const float*)d_in[8];
    const float* W_UV   = (const float*)d_in[9];  const float* b_UV  = (const float*)d_in[10];
    const float* W_UQ   = (const float*)d_in[11]; const float* b_UQ  = (const float*)d_in[12];
    const float* W_KR   = (const float*)d_in[13]; const float* b_KR  = (const float*)d_in[14];
    const float* W_QR   = (const float*)d_in[15]; const float* b_QR  = (const float*)d_in[16];
    const float* W_O    = (const float*)d_in[17]; const float* b_O   = (const float*)d_in[18];
    float* out = (float*)d_out;

    // workspace carve-out (f16 everywhere), ~210 MB total
    char* ws = (char*)d_ws;
    size_t off = 0;
    auto alloc = [&](size_t elems) -> _Float16* {
        _Float16* p = (_Float16*)(ws + off);
        off += ((elems * sizeof(_Float16)) + 255) & ~(size_t)255;
        return p;
    };
    const int M = B_ * S_;
    _Float16* hf    = alloc((size_t)M * HID_);
    _Float16* tDKV  = alloc((size_t)HID_ * CKV_);
    _Float16* tDQ   = alloc((size_t)HID_ * CQ_);
    _Float16* tUK   = alloc((size_t)CKV_ * H_ * D_);
    _Float16* tUV   = alloc((size_t)CKV_ * H_ * D_);
    _Float16* tUQ   = alloc((size_t)CQ_ * H_ * D_);
    _Float16* tKR   = alloc((size_t)HID_ * H_ * R_);
    _Float16* tQR   = alloc((size_t)CQ_ * H_ * R_);
    _Float16* tO    = alloc((size_t)HID_ * HID_);
    _Float16* ckv   = alloc((size_t)M * CKV_);
    _Float16* cq    = alloc((size_t)M * CQ_);
    _Float16* qlat  = alloc((size_t)M * H_ * D_);
    _Float16* qrope = alloc((size_t)M * H_ * R_);
    _Float16* klat  = alloc((size_t)M * H_ * D_);
    _Float16* krope = alloc((size_t)M * H_ * R_);
    _Float16* vlin  = alloc((size_t)M * H_ * D_);
    _Float16* Qp    = alloc((size_t)M * H_ * 192);
    _Float16* Kp    = alloc((size_t)M * H_ * 192);
    _Float16* Vp    = alloc((size_t)M * H_ * D_);
    _Float16* aout  = alloc((size_t)M * H_ * D_);

    // --- operand prep: A to f16, weights to f16 [N,K] ----------------------
    {
        int n = M * HID_;
        f32_to_f16<<<(n + 255) / 256, 256, 0, stream>>>(hidden, hf, n);
    }
    auto tr = [&](const float* W, _Float16* Wt, int K, int N) {
        transpose_f32_f16<<<dim3(N / 32, K / 32), 256, 0, stream>>>(W, Wt, K, N);
    };
    tr(W_DKV, tDKV, HID_, CKV_);
    tr(W_DQ,  tDQ,  HID_, CQ_);
    tr(W_UK,  tUK,  CKV_, H_ * D_);
    tr(W_UV,  tUV,  CKV_, H_ * D_);
    tr(W_UQ,  tUQ,  CQ_,  H_ * D_);
    tr(W_KR,  tKR,  HID_, H_ * R_);
    tr(W_QR,  tQR,  CQ_,  H_ * R_);
    tr(W_O,   tO,   HID_, HID_);

    // --- projection GEMMs (TDM-fed WMMA) -----------------------------------
    dim3 blk(256);
    gemm_wmma<_Float16><<<dim3(CKV_ / 64,    M / 128), blk, 0, stream>>>(hf,  tDKV, b_DKV, ckv,   M, CKV_,    HID_);
    gemm_wmma<_Float16><<<dim3(H_ * R_ / 64, M / 128), blk, 0, stream>>>(hf,  tKR,  b_KR,  krope, M, H_ * R_, HID_);
    gemm_wmma<_Float16><<<dim3(CQ_ / 64,     M / 128), blk, 0, stream>>>(hf,  tDQ,  b_DQ,  cq,    M, CQ_,     HID_);
    gemm_wmma<_Float16><<<dim3(H_ * D_ / 64, M / 128), blk, 0, stream>>>(ckv, tUK,  b_UK,  klat,  M, H_ * D_, CKV_);
    gemm_wmma<_Float16><<<dim3(H_ * D_ / 64, M / 128), blk, 0, stream>>>(ckv, tUV,  b_UV,  vlin,  M, H_ * D_, CKV_);
    gemm_wmma<_Float16><<<dim3(H_ * D_ / 64, M / 128), blk, 0, stream>>>(cq,  tUQ,  b_UQ,  qlat,  M, H_ * D_, CQ_);
    gemm_wmma<_Float16><<<dim3(H_ * R_ / 64, M / 128), blk, 0, stream>>>(cq,  tQR,  b_QR,  qrope, M, H_ * R_, CQ_);

    rope_pack<<<dim3(M), dim3(256), 0, stream>>>(qlat, qrope, klat, krope, vlin,
                                                 pos, Qp, Kp, Vp);

    flash_attn<<<dim3(S_ / 64, H_, B_), dim3(128), 0, stream>>>(Qp, Kp, Vp, mask, aout);

    gemm_wmma<float><<<dim3(HID_ / 64, M / 128), blk, 0, stream>>>(aout, tO, b_O, out,
                                                                   M, HID_, HID_);
}
